// EdgeAttention_41068477284523
// MI455X (gfx1250) — matrix operand order repaired and checked by reading, not compiled, not run on hardware
//
#include <hip/hip_runtime.h>

typedef __attribute__((ext_vector_type(16))) _Float16 v16h;
typedef __attribute__((ext_vector_type(8)))  _Float16 v8h;
typedef __attribute__((ext_vector_type(4)))  _Float16 v4h;
typedef __attribute__((ext_vector_type(8)))  float    v8f;
typedef __attribute__((ext_vector_type(4)))  float    v4f;

#define B_ 2
#define N_ 24
#define E_ 128
#define H_ 8
#define D_ 16
#define M_ (B_ * N_ * N_)   /* 1152 rows for the projection GEMMs */
#define W_ELEMS (E_ * E_)   /* 16384 per weight matrix */

#define CAT16 0, 1, 2, 3, 4, 5, 6, 7, 8, 9, 10, 11, 12, 13, 14, 15

// ---------------------------------------------------------------------------
// Kernel 0: one-time f32 -> f16 conversion of activations and weights.
// grid = (144, 8): y = segment (0-2 activations of 147456, 3-7 weights of 16384)
// ---------------------------------------------------------------------------
__global__ void convert_kernel(const float* __restrict__ q, const float* __restrict__ kt,
                               const float* __restrict__ v,
                               const float* __restrict__ w0, const float* __restrict__ w1,
                               const float* __restrict__ w2, const float* __restrict__ w3,
                               const float* __restrict__ w4,
                               _Float16* __restrict__ q16, _Float16* __restrict__ kt16,
                               _Float16* __restrict__ v16, _Float16* __restrict__ w16) {
  const float* src; _Float16* dst; int n;
  switch (blockIdx.y) {
    case 0: src = q;  dst = q16;  n = M_ * E_; break;
    case 1: src = kt; dst = kt16; n = M_ * E_; break;
    case 2: src = v;  dst = v16;  n = M_ * E_; break;
    case 3: src = w0; dst = w16 + 0 * W_ELEMS; n = W_ELEMS; break;
    case 4: src = w1; dst = w16 + 1 * W_ELEMS; n = W_ELEMS; break;
    case 5: src = w2; dst = w16 + 2 * W_ELEMS; n = W_ELEMS; break;
    case 6: src = w3; dst = w16 + 3 * W_ELEMS; n = W_ELEMS; break;
    default: src = w4; dst = w16 + 4 * W_ELEMS; n = W_ELEMS; break;
  }
  const int i = (blockIdx.x * 256 + threadIdx.x) * 4;
  if (i >= n) return;
  v4f x = *(const v4f*)(src + i);
  v4h y;
#pragma unroll
  for (int j = 0; j < 4; ++j) y[j] = (_Float16)x[j];
  *(v4h*)(dst + i) = y;
}

// ---------------------------------------------------------------------------
// GEMM tile: out[m,n] = sum_k A[m,k] * W[n,k]  (A, W in f16; K = 128)
// Block = 256 threads = 8 waves; blockIdx.x = 16-row M tile; wave = 16-col N tile.
// A tile staged into LDS with GLOBAL_LOAD_ASYNC_TO_LDS_B128 (ASYNCcnt-tracked
// DMA, no VGPR round-trip), then fragments built with ds_load_b128.
// Per ISA 7.12.2 (16-bit A): lane m=lane&15, K-run = (lane&16 ? 8 : 0) for
// elems 0-7 and +16 for elems 8-15 -> two contiguous 8xf16 chunks.
// B (32x16, B[k][n] = W[n][k]): lane n=lane&15, K-run = (lane&16 ? 8 : 0), +16 for elems 8-15,
// two contiguous 8xf16 chunks in W's row n (same k order as A).
// ---------------------------------------------------------------------------
template <typename OutT>
__device__ __forceinline__ void gemm_xwt_tile(const _Float16* __restrict__ A16,
                                              const _Float16* __restrict__ W16,
                                              OutT* __restrict__ out,
                                              _Float16* __restrict__ As) {
  const int lane = threadIdx.x & 31;
  const int wave = threadIdx.x >> 5;
  const int m0 = blockIdx.x * 16;

  // async DMA stage of the 16x128 f16 A tile (4 KB): 16 contiguous bytes/thread
  {
    const uint32_t laddr = (uint32_t)(uintptr_t)(&As[threadIdx.x * 8]);
    const uint64_t gaddr = (uint64_t)(uintptr_t)(&A16[m0 * E_ + threadIdx.x * 8]);
    asm volatile("global_load_async_to_lds_b128 %0, %1, off"
                 :: "v"(laddr), "v"(gaddr) : "memory");
    asm volatile("s_wait_asynccnt 0x0" ::: "memory");
  }
  __syncthreads();

  const int n0  = wave * 16;
  const int m   = lane & 15;
  const int kbA = (lane & 16) ? 8 : 0;
  const int kbB = (lane & 16) ? 8 : 0;
  const _Float16* wrow = W16 + (n0 + (lane & 15)) * E_ + kbB;

  v8f c = {};
#pragma unroll
  for (int k0 = 0; k0 < E_; k0 += 32) {
    v8h alo = *(const v8h*)&As[m * E_ + k0 + kbA];
    v8h ahi = *(const v8h*)&As[m * E_ + k0 + 16 + kbA];
    v16h a = __builtin_shufflevector(alo, ahi, CAT16);
    v8h blo = *(const v8h*)(wrow + k0);
    v8h bhi = *(const v8h*)(wrow + k0 + 16);
    v16h b = __builtin_shufflevector(blo, bhi, CAT16);
    c = __builtin_amdgcn_wmma_f32_16x16x32_f16(false, a, false, b, (short)0, c, false, false);
  }
  // C/D layout: VGPR j -> M = j (+8 for lanes 16-31), N = lane&15
  const int mo = m0 + ((lane & 16) ? 8 : 0);
  const int nn = n0 + (lane & 15);
#pragma unroll
  for (int j = 0; j < 8; ++j)
    out[(mo + j) * E_ + nn] = (OutT)c[j];
}

// ---------------------------------------------------------------------------
// Kernel 1: four input projections in one launch (grid.z selects which).
// lk/rk stored f16 (feed WMMA scores); lv/rv stored f32 (feed f32 combine).
// ---------------------------------------------------------------------------
__global__ void proj4_kernel(const _Float16* __restrict__ q16, const _Float16* __restrict__ kt16,
                             const _Float16* __restrict__ v16, const _Float16* __restrict__ w16,
                             _Float16* __restrict__ lk16, _Float16* __restrict__ rk16,
                             float* __restrict__ lv, float* __restrict__ rv) {
  __shared__ __align__(16) _Float16 As[16 * E_];
  switch (blockIdx.z) {
    case 0:  gemm_xwt_tile<_Float16>(q16,  w16 + 0 * W_ELEMS, lk16, As); break;
    case 1:  gemm_xwt_tile<_Float16>(kt16, w16 + 1 * W_ELEMS, rk16, As); break;
    case 2:  gemm_xwt_tile<float>   (v16,  w16 + 2 * W_ELEMS, lv,   As); break;
    default: gemm_xwt_tile<float>   (v16,  w16 + 3 * W_ELEMS, rv,   As); break;
  }
}

// ---------------------------------------------------------------------------
// Kernel 2: scores[b,h,x,a,y] = 0.25 * sum_d lk[b,x,a,h,d] * rk[b,a,y,h,d]
// One wave per (b,h,a): 24x24x16 GEMM as 2x2 WMMA tiles, K zero-padded 16->32.
// ---------------------------------------------------------------------------
__global__ void scores_kernel(const _Float16* __restrict__ lk16,
                              const _Float16* __restrict__ rk16,
                              float* __restrict__ scores) {
  const int lane = threadIdx.x & 31;
  const int g = (blockIdx.x << 3) + (threadIdx.x >> 5);  // 384 waves == B_*H_*N_
  const int a = g % N_;
  const int h = (g / N_) % H_;
  const int b = g / (N_ * H_);

  const _Float16* abase = lk16 + ((b * N_) * N_ + a) * E_ + h * D_;  // x stride N_*E_
  const _Float16* bbase = rk16 + ((b * N_ + a) * N_) * E_ + h * D_;  // y stride E_
  float* obase = scores + (((b * H_ + h) * N_) * N_ + a) * N_;       // + x*N_*N_ + y

  const v8h z8 = {};
#pragma unroll
  for (int mt = 0; mt < 2; ++mt) {
    const int m = mt * 16 + (lane & 15);
    const int kbA = (lane & 16) ? 8 : 0;
    v8h alo = (m < N_) ? *(const v8h*)&abase[m * (N_ * E_) + kbA] : z8;
    v16h af = __builtin_shufflevector(alo, z8, CAT16);   // K 16..31 = zero pad
#pragma unroll
    for (int nt = 0; nt < 2; ++nt) {
      const int n = nt * 16 + (lane & 15);
      v16h bf = {};
      if (n < N_) {                                      // elems 8-15 carry K>=16: pad
        v8h blo = *(const v8h*)&bbase[n * E_ + ((lane & 16) ? 8 : 0)];
        bf = __builtin_shufflevector(blo, z8, CAT16);
      }
      v8f c = {};
      c = __builtin_amdgcn_wmma_f32_16x16x32_f16(false, af, false, bf, (short)0, c, false, false);
      const int x0 = mt * 16 + ((lane & 16) ? 8 : 0);
      const int y  = nt * 16 + (lane & 15);
      if (y < N_) {
#pragma unroll
        for (int j = 0; j < 8; ++j) {
          const int x = x0 + j;
          if (x < N_) obase[x * N_ * N_ + y] = 0.25f * c[j];
        }
      }
    }
  }
}

// ---------------------------------------------------------------------------
// Kernel 3: softmax over a (f32) + xout[b,x,y,h,d] = sum_a att*lv*rv (f32),
// stored as f16 to feed the final WMMA projection.
// ---------------------------------------------------------------------------
__global__ void softmax_combine_kernel(const float* __restrict__ scores,
                                       const float* __restrict__ lv,
                                       const float* __restrict__ rv,
                                       _Float16* __restrict__ xo16) {
  const int idx = blockIdx.x * blockDim.x + threadIdx.x;   // (b,x,y,h), 9216 total
  const int h = idx & (H_ - 1);
  const int y = (idx >> 3) % N_;
  const int x = (idx / (H_ * N_)) % N_;
  const int b = idx / (H_ * N_ * N_);

  const float* sp = scores + (((b * H_ + h) * N_ + x) * N_) * N_ + y;  // stride N_ over a
  float s[N_];
  float mx = -3.0e38f;
#pragma unroll
  for (int a = 0; a < N_; ++a) { s[a] = sp[a * N_]; mx = fmaxf(mx, s[a]); }
  float sum = 0.0f;
#pragma unroll
  for (int a = 0; a < N_; ++a) { s[a] = __expf(s[a] - mx); sum += s[a]; }
  const float inv = 1.0f / sum;

  v4f acc[D_ / 4];
#pragma unroll
  for (int j = 0; j < D_ / 4; ++j) acc[j] = (v4f){};

  const float* lvb = lv + ((b * N_ + x) * N_) * E_ + h * D_;   // + a*E_
  const float* rvb = rv + ((b * N_) * N_ + y) * E_ + h * D_;   // + a*N_*E_
  for (int a = 0; a < N_; ++a) {
    const float w = s[a] * inv;
    const v4f* lva = (const v4f*)(lvb + a * E_);        // 64B-aligned (h*16 floats)
    const v4f* rva = (const v4f*)(rvb + a * N_ * E_);
#pragma unroll
    for (int q = 0; q < D_ / 4; ++q) {
      v4f lx = lva[q];
      v4f rx = rva[q];
#pragma unroll
      for (int j = 0; j < 4; ++j) acc[q][j] += w * lx[j] * rx[j];
    }
  }
  _Float16* op = xo16 + ((b * N_ + x) * N_ + y) * E_ + h * D_;
#pragma unroll
  for (int q = 0; q < D_ / 4; ++q) {
    v4h yv;
#pragma unroll
    for (int j = 0; j < 4; ++j) yv[j] = (_Float16)acc[q][j];
    *(v4h*)(op + q * 4) = yv;
  }
}

// ---------------------------------------------------------------------------
// Kernel 4: final projection out = xout @ W_out.T (f16 in, f32 out -> d_out)
// ---------------------------------------------------------------------------
__global__ void outproj_kernel(const _Float16* __restrict__ xo16,
                               const _Float16* __restrict__ wout16,
                               float* __restrict__ out) {
  __shared__ __align__(16) _Float16 As[16 * E_];
  gemm_xwt_tile<float>(xo16, wout16, out, As);
}

extern "C" void kernel_launch(void* const* d_in, const int* in_sizes, int n_in,
                              void* d_out, int out_size, void* d_ws, size_t ws_size,
                              hipStream_t stream) {
  (void)in_sizes; (void)n_in; (void)out_size; (void)ws_size;
  const float* query = (const float*)d_in[0];
  const float* key_t = (const float*)d_in[1];
  const float* value = (const float*)d_in[2];
  const float* W_lk  = (const float*)d_in[3];
  const float* W_rk  = (const float*)d_in[4];
  const float* W_lv  = (const float*)d_in[5];
  const float* W_rv  = (const float*)d_in[6];
  const float* W_out = (const float*)d_in[7];
  float* out = (float*)d_out;

  const size_t T = (size_t)M_ * E_;   // 147456 elements per projected tensor
  char* p = (char*)d_ws;
  float*    lv   = (float*)p;    p += T * 4;
  float*    rv   = (float*)p;    p += T * 4;
  float*    sc   = (float*)p;    p += (size_t)B_ * H_ * N_ * N_ * N_ * 4;  // 221184 f32
  _Float16* q16  = (_Float16*)p; p += T * 2;
  _Float16* kt16 = (_Float16*)p; p += T * 2;
  _Float16* v16  = (_Float16*)p; p += T * 2;
  _Float16* lk16 = (_Float16*)p; p += T * 2;
  _Float16* rk16 = (_Float16*)p; p += T * 2;
  _Float16* xo16 = (_Float16*)p; p += T * 2;
  _Float16* w16  = (_Float16*)p; p += (size_t)5 * W_ELEMS * 2;

  const dim3 blk(256, 1, 1);
  convert_kernel<<<dim3((M_ * E_) / 1024, 8, 1), blk, 0, stream>>>(
      query, key_t, value, W_lk, W_rk, W_lv, W_rv, W_out, q16, kt16, v16, w16);
  proj4_kernel<<<dim3(M_ / 16, 1, 4), blk, 0, stream>>>(
      q16, kt16, v16, w16, lk16, rk16, lv, rv);
  scores_kernel<<<dim3((B_ * H_ * N_) / 8, 1, 1), blk, 0, stream>>>(lk16, rk16, sc);
  softmax_combine_kernel<<<dim3((B_ * N_ * N_ * H_) / 256, 1, 1), blk, 0, stream>>>(
      sc, lv, rv, xo16);
  outproj_kernel<<<dim3(M_ / 16, 1, 1), blk, 0, stream>>>(xo16, w16 + 4 * W_ELEMS, out);
}
